// GPQSoftMaxNet_90683939488057
// MI455X (gfx1250) — matrix operand order repaired and checked by reading, not compiled, not run on hardware
//
#include <hip/hip_runtime.h>

// GEMM: logits[M,NC] = (features[M,K] @ Prototypes[K,NC]) / 16
// M=32768, K=256, NC=4096.  bf16x3 split-precision WMMA (fp32-accurate).

typedef __attribute__((ext_vector_type(4)))  float  v4f;
typedef __attribute__((ext_vector_type(8)))  float  v8f;
typedef __attribute__((ext_vector_type(4)))  __bf16 v4bf;
typedef __attribute__((ext_vector_type(8)))  __bf16 v8bf;
typedef __attribute__((ext_vector_type(16))) __bf16 v16bf;

#define KDIM 256
#define BM   128
#define BN   128
#define KC   32
#define LDW  40   // padded LDS row stride (elements) for one 32-wide K slice

__device__ __forceinline__ void cvt_split8(const float* __restrict__ x, v8bf& hi, v8bf& lo) {
#pragma unroll
  for (int e = 0; e < 8; ++e) {
    float f  = x[e];
    __bf16 h = (__bf16)f;
    hi[e] = h;
    lo[e] = (__bf16)(f - (float)h);
  }
}

__global__ __launch_bounds__(256)
void gpq_logits_wmma_bf16x3(const float* __restrict__ A,
                            const float* __restrict__ B,
                            float* __restrict__ C,
                            int NC) {
  __shared__ __align__(16) __bf16 sAh[BM * LDW];
  __shared__ __align__(16) __bf16 sAl[BM * LDW];
  __shared__ __align__(16) __bf16 sBh[BN * LDW];
  __shared__ __align__(16) __bf16 sBl[BN * LDW];

  const int tid      = threadIdx.x;
  const int lane     = tid & 31;
  const int wave     = tid >> 5;
  const int waveM    = wave >> 2;      // 0..1  (64-row strip)
  const int waveN    = wave & 3;       // 0..3  (32-col strip)
  const int laneRow  = lane & 15;
  const int laneHalf = lane >> 4;      // 0 or 1

  // ---- global->LDS staging mapping ----
  const int rowA = tid >> 1;           // 0..127
  const int colA = (tid & 1) << 4;     // 0 or 16 (k within slice)
  const int kbB  = tid >> 5;           // 0..7   (k block of 4)
  const int cbB  = tid & 31;           // 0..31  (col block of 4)

  const float* gA = A + (size_t)(blockIdx.y * BM + rowA) * KDIM + colA;
  const float* gB = B + (size_t)blockIdx.x * BN + cbB * 4;

  // prologue: fetch K-slice 0 into registers
  v4f ra[4], rb[4];
#pragma unroll
  for (int j = 0; j < 4; ++j) ra[j] = *(const v4f*)(gA + j * 4);
#pragma unroll
  for (int i = 0; i < 4; ++i) rb[i] = *(const v4f*)(gB + (size_t)(kbB * 4 + i) * NC);

  v8f acc[4][2];
#pragma unroll
  for (int i = 0; i < 4; ++i)
#pragma unroll
    for (int j = 0; j < 2; ++j)
#pragma unroll
      for (int e = 0; e < 8; ++e) acc[i][j][e] = 0.0f;

  const int numK = KDIM / KC;          // 8
  for (int kt = 0; kt < numK; ++kt) {
    __syncthreads();
    {
      // --- A slice: 16 consecutive k of one row -> hi/lo bf16, row-major ---
      float av[16];
#pragma unroll
      for (int j = 0; j < 4; ++j)
#pragma unroll
        for (int e = 0; e < 4; ++e) av[j * 4 + e] = ra[j][e];
      v8bf h0, l0, h1, l1;
      cvt_split8(av,     h0, l0);
      cvt_split8(av + 8, h1, l1);
      const int ab = rowA * LDW + colA;
      *(v8bf*)(sAh + ab)     = h0;
      *(v8bf*)(sAh + ab + 8) = h1;
      *(v8bf*)(sAl + ab)     = l0;
      *(v8bf*)(sAl + ab + 8) = l1;

      // --- B slice: 4x4 (k x col) register transpose -> LDS [col][k] ---
#pragma unroll
      for (int j = 0; j < 4; ++j) {
        float cv[4];
#pragma unroll
        for (int e = 0; e < 4; ++e) cv[e] = rb[e][j];
        v4bf bh, bl;
#pragma unroll
        for (int e = 0; e < 4; ++e) {
          __bf16 h = (__bf16)cv[e];
          bh[e] = h;
          bl[e] = (__bf16)(cv[e] - (float)h);
        }
        const int bb = (cbB * 4 + j) * LDW + kbB * 4;
        *(v4bf*)(sBh + bb) = bh;
        *(v4bf*)(sBl + bb) = bl;
      }
    }
    __syncthreads();

    // prefetch next K-slice from HBM while WMMAs run on LDS data
    if (kt + 1 < numK) {
      const int kn = (kt + 1) * KC;
#pragma unroll
      for (int j = 0; j < 4; ++j) ra[j] = *(const v4f*)(gA + kn + j * 4);
#pragma unroll
      for (int i = 0; i < 4; ++i)
        rb[i] = *(const v4f*)(gB + (size_t)(kn + kbB * 4 + i) * NC);
    }

    // --- fragment loads (ISA 16-bit A/B VGPR layouts) ---
    v16bf ah[4], al[4], bh[2], bl[2];
#pragma unroll
    for (int i = 0; i < 4; ++i) {
      const int ab = (waveM * 64 + i * 16 + laneRow) * LDW + laneHalf * 8;
      v8bf p0 = *(const v8bf*)(sAh + ab);
      v8bf p1 = *(const v8bf*)(sAh + ab + 16);
      ah[i] = __builtin_shufflevector(p0, p1, 0,1,2,3,4,5,6,7,8,9,10,11,12,13,14,15);
      v8bf q0 = *(const v8bf*)(sAl + ab);
      v8bf q1 = *(const v8bf*)(sAl + ab + 16);
      al[i] = __builtin_shufflevector(q0, q1, 0,1,2,3,4,5,6,7,8,9,10,11,12,13,14,15);
    }
#pragma unroll
    for (int j = 0; j < 2; ++j) {
      const int bb = (waveN * 32 + j * 16 + laneRow) * LDW + laneHalf * 8;
      v8bf p0 = *(const v8bf*)(sBh + bb);
      v8bf p1 = *(const v8bf*)(sBh + bb + 16);
      bh[j] = __builtin_shufflevector(p0, p1, 0,1,2,3,4,5,6,7,8,9,10,11,12,13,14,15);
      v8bf q0 = *(const v8bf*)(sBl + bb);
      v8bf q1 = *(const v8bf*)(sBl + bb + 16);
      bl[j] = __builtin_shufflevector(q0, q1, 0,1,2,3,4,5,6,7,8,9,10,11,12,13,14,15);
    }

    // --- 8 tiles x 3 split-precision WMMAs ---
#pragma unroll
    for (int i = 0; i < 4; ++i)
#pragma unroll
      for (int j = 0; j < 2; ++j) {
        acc[i][j] = __builtin_amdgcn_wmma_f32_16x16x32_bf16(
            false, ah[i], false, bh[j], (short)0, acc[i][j], false, false);
        acc[i][j] = __builtin_amdgcn_wmma_f32_16x16x32_bf16(
            false, ah[i], false, bl[j], (short)0, acc[i][j], false, false);
        acc[i][j] = __builtin_amdgcn_wmma_f32_16x16x32_bf16(
            false, al[i], false, bh[j], (short)0, acc[i][j], false, false);
      }
  }

  // --- epilogue: scale by 1/n_book and stream out (non-temporal: C >> L2) ---
  const size_t crow0 = (size_t)blockIdx.y * BM + waveM * 64 + laneHalf * 8;
  const int    ccol0 = blockIdx.x * BN + waveN * 32 + laneRow;
#pragma unroll
  for (int i = 0; i < 4; ++i)
#pragma unroll
    for (int j = 0; j < 2; ++j)
#pragma unroll
      for (int r = 0; r < 8; ++r) {
        const size_t row = crow0 + (size_t)i * 16 + r;
        const int    col = ccol0 + j * 16;
        __builtin_nontemporal_store(acc[i][j][r] * 0.0625f, &C[row * (size_t)NC + col]);
      }
}

extern "C" void kernel_launch(void* const* d_in, const int* in_sizes, int n_in,
                              void* d_out, int out_size, void* d_ws, size_t ws_size,
                              hipStream_t stream) {
  const float* A = (const float*)d_in[0];   // features  [M, 256]
  const float* B = (const float*)d_in[1];   // Prototypes[256, NC]
  float* C = (float*)d_out;                 // logits    [M, NC]
  const int M  = in_sizes[0] / KDIM;        // 32768
  const int NC = in_sizes[1] / KDIM;        // 4096
  dim3 grid(NC / BN, M / BM);               // (32, 256)
  gpq_logits_wmma_bf16x3<<<grid, dim3(256), 0, stream>>>(A, B, C, NC);
}
